// ShiftAwareAttention1D_10720238370967
// MI455X (gfx1250) — matrix operand order repaired and checked by reading, not compiled, not run on hardware
//
#include <hip/hip_runtime.h>
#include <hip/hip_bf16.h>

// ---------------------------------------------------------------------------
// ShiftAwareAttention1D for MI455X (gfx1250, wave32, WMMA)
//
// Pipeline (fp32 tensors, f16 WMMA w/ f32 accumulate for the GEMMs):
//   1. vproj  = value @ Wv.T   + bv    (16384 x 1024 x 1024)  WMMA 128x128 tiles
//   2. poff   = query @ Woff.T + boff  (16384 x   64 x 1024)  WMMA 256x64 tiles
//   3. plogit = query @ Ww.T   + bw    (16384 x   64 x 1024)  WMMA 256x64 tiles
//   4. sampler: softmax over P, faithful grid_sample(align_corners=True) on
//      input shaped (B*H, d_k, H=L, W=1); W==1 makes the x-lerp collapse
//      exactly, leaving a 2-tap y-lerp.
//   5. out    = agg @ Wout.T + bout    (16384 x 1024 x 1024)  WMMA 128x128 tiles
//
// GEMM inner loop is software-pipelined: next K-tile global loads are issued
// before the current tile's WMMAs; convert+LDS-store happen after, so the
// single s_wait_loadcnt sits behind the matrix ops instead of between loads.
// ---------------------------------------------------------------------------

#define BDIM   8
#define LDIM   2048
#define DDIM   1024
#define NHEAD  16
#define NPOINT 4
#define HP     (NHEAD * NPOINT)   // 64
#define MROWS  (BDIM * LDIM)      // 16384

typedef __attribute__((ext_vector_type(16))) _Float16     v16h;
typedef __attribute__((ext_vector_type(8)))  float        v8f;
typedef __attribute__((ext_vector_type(4)))  _Float16     v4h;
typedef __attribute__((ext_vector_type(4)))  unsigned int v4u;
typedef __attribute__((ext_vector_type(4)))  float        v4f;

#define BK   32
#define LSTR 40   // LDS row stride in halfs: 32 + 8 pad (80B = 5*16B, aligned)

union HFrag { v16h h; v4u u[2]; };

// ---------------- WMMA GEMM: C[M,N] = A[M,K] * W[N,K]^T + bias[N] ----------
// 256 threads = 8 waves. Wave tile = (WMT*16) x (WNT*16). Double-buffered LDS.
template<int BM, int BN, int WMT, int WNT>
__global__ __launch_bounds__(256)
void wmma_gemm_bias(const float* __restrict__ A,
                    const float* __restrict__ W,
                    const float* __restrict__ bias,
                    float* __restrict__ C,
                    int M, int N, int K)
{
  constexpr int NWM = BM / (WMT * 16);   // waves along M
  constexpr int NWN = BN / (WNT * 16);   // waves along N
  static_assert(NWM * NWN == 8, "need exactly 8 waves");
  constexpr int APASS = BM / 32;         // float4 staging passes for A
  constexpr int BPASS = BN / 32;         // float4 staging passes for B

  __shared__ _Float16 As[2 * BM * LSTR];
  __shared__ _Float16 Bs[2 * BN * LSTR];

  const int tid   = threadIdx.x;
  const int lane  = tid & 31;
  const int wid   = tid >> 5;
  const int waveM = wid % NWM;
  const int waveN = wid / NWM;
  const int lrow  = lane & 15;
  const int lhalf = lane >> 4;

  const int m0 = blockIdx.x * BM;
  const int n0 = blockIdx.y * BN;
  const int wm0 = waveM * (WMT * 16);
  const int wn0 = waveN * (WNT * 16);

  // staging pattern: 8 threads/row, float4 each -> 32 rows x 32 cols per pass
  const int srow  = tid >> 3;
  const int scol4 = (tid & 7) * 4;

  v8f acc[WMT][WNT] = {};
  v4f aReg[APASS], bReg[BPASS];

  // ---- prologue: fetch + stage K-tile 0 into buffer 0 ----
  #pragma unroll
  for (int rr = 0; rr < APASS; ++rr)
    aReg[rr] = *(const v4f*)(&A[(size_t)(m0 + srow + rr * 32) * K + scol4]);
  #pragma unroll
  for (int rr = 0; rr < BPASS; ++rr)
    bReg[rr] = *(const v4f*)(&W[(size_t)(n0 + srow + rr * 32) * K + scol4]);
  #pragma unroll
  for (int rr = 0; rr < APASS; ++rr) {
    v4f f = aReg[rr];
    v4h h = { (_Float16)f.x, (_Float16)f.y, (_Float16)f.z, (_Float16)f.w };
    *(v4h*)(&As[(srow + rr * 32) * LSTR + scol4]) = h;
  }
  #pragma unroll
  for (int rr = 0; rr < BPASS; ++rr) {
    v4f f = bReg[rr];
    v4h h = { (_Float16)f.x, (_Float16)f.y, (_Float16)f.z, (_Float16)f.w };
    *(v4h*)(&Bs[(srow + rr * 32) * LSTR + scol4]) = h;
  }
  __syncthreads();

  for (int k0 = 0; k0 < K; k0 += BK) {
    const int  cur  = (k0 / BK) & 1;
    const int  nxt  = cur ^ 1;
    const bool more = (k0 + BK) < K;

    // ---- issue next K-tile global loads early (consumed after WMMAs) ----
    if (more) {
      #pragma unroll
      for (int rr = 0; rr < APASS; ++rr)
        aReg[rr] = *(const v4f*)(&A[(size_t)(m0 + srow + rr * 32) * K + k0 + BK + scol4]);
      #pragma unroll
      for (int rr = 0; rr < BPASS; ++rr)
        bReg[rr] = *(const v4f*)(&W[(size_t)(n0 + srow + rr * 32) * K + k0 + BK + scol4]);
      if (k0 + 2 * BK < K)
        __builtin_prefetch(&A[(size_t)(m0 + srow) * K + k0 + 2 * BK + scol4], 0, 1);
    }

    // ---- fragments from current buffer ----
    const _Float16* Ab = &As[cur * BM * LSTR];
    const _Float16* Bb = &Bs[cur * BN * LSTR];
    HFrag a[WMT], b[WNT];
    // A 16x32 f16 (ISA 7.12.2): half 0 -> K[0..7],K[16..23]; half 1 -> K[8..15],K[24..31]
    #pragma unroll
    for (int tm = 0; tm < WMT; ++tm) {
      const int row = wm0 + tm * 16 + lrow;
      a[tm].u[0] = *(const v4u*)(&Ab[row * LSTR + lhalf * 8]);
      a[tm].u[1] = *(const v4u*)(&Ab[row * LSTR + 16 + lhalf * 8]);
    }
    // B 32x16 f16: half 0 -> K[0..7],K[16..23]; half 1 -> K[8..15],K[24..31]
    #pragma unroll
    for (int tn = 0; tn < WNT; ++tn) {
      const int col = wn0 + tn * 16 + lrow;
      b[tn].u[0] = *(const v4u*)(&Bb[col * LSTR + lhalf * 8]);
      b[tn].u[1] = *(const v4u*)(&Bb[col * LSTR + 16 + lhalf * 8]);
    }

    #pragma unroll
    for (int tm = 0; tm < WMT; ++tm)
      #pragma unroll
      for (int tn = 0; tn < WNT; ++tn)
        acc[tm][tn] = __builtin_amdgcn_wmma_f32_16x16x32_f16(
            false, a[tm].h, false, b[tn].h,
            (short)0, acc[tm][tn], false, false);

    // ---- convert + stage next tile into the other buffer ----
    if (more) {
      #pragma unroll
      for (int rr = 0; rr < APASS; ++rr) {
        v4f f = aReg[rr];
        v4h h = { (_Float16)f.x, (_Float16)f.y, (_Float16)f.z, (_Float16)f.w };
        *(v4h*)(&As[nxt * BM * LSTR + (srow + rr * 32) * LSTR + scol4]) = h;
      }
      #pragma unroll
      for (int rr = 0; rr < BPASS; ++rr) {
        v4f f = bReg[rr];
        v4h h = { (_Float16)f.x, (_Float16)f.y, (_Float16)f.z, (_Float16)f.w };
        *(v4h*)(&Bs[nxt * BN * LSTR + (srow + rr * 32) * LSTR + scol4]) = h;
      }
    }
    __syncthreads();   // covers RAW on nxt and WAR on cur for next iteration
  }

  // ---- epilogue: C/D layout = col lane&15, rows r + 8*(lane>>4) ----
  #pragma unroll
  for (int tm = 0; tm < WMT; ++tm) {
    #pragma unroll
    for (int tn = 0; tn < WNT; ++tn) {
      const int col = n0 + wn0 + tn * 16 + lrow;
      const float bv = bias[col];
      #pragma unroll
      for (int r = 0; r < 8; ++r) {
        const int row = m0 + wm0 + tm * 16 + r + 8 * lhalf;
        C[(size_t)row * N + col] = acc[tm][tn][r] + bv;
      }
    }
  }
}

// ---------------- sampler: softmax + faithful grid_sample + aggregate ------
__global__ __launch_bounds__(256)
void sample_agg_kernel(const float* __restrict__ vproj,   // (B, L, D)
                       const float* __restrict__ poff,    // (B, L, HP)
                       const float* __restrict__ plogit,  // (B, L, HP)
                       float* __restrict__ agg)           // (B, L, D)
{
  const int bl  = blockIdx.x;          // 0 .. B*L-1
  const int b   = bl / LDIM;
  const int l   = bl % LDIM;
  const int tid = threadIdx.x;

  __shared__ float s_w [NHEAD][NPOINT];
  __shared__ int   s_y0[NHEAD][NPOINT];
  __shared__ float s_wy[NHEAD][NPOINT];

  if (tid < NHEAD) {
    const int h = tid;
    float off[NPOINT], lg[NPOINT];
    #pragma unroll
    for (int p = 0; p < NPOINT; ++p) {
      off[p] = poff  [(size_t)bl * HP + h * NPOINT + p];
      lg[p]  = plogit[(size_t)bl * HP + h * NPOINT + p];
    }
    float mx = lg[0];
    #pragma unroll
    for (int p = 1; p < NPOINT; ++p) mx = fmaxf(mx, lg[p]);
    float e[NPOINT], sum = 0.f;
    #pragma unroll
    for (int p = 0; p < NPOINT; ++p) { e[p] = __expf(lg[p] - mx); sum += e[p]; }
    const float inv = 1.f / sum;

    // faithful grid_sample(align_corners=True) on (.., H=L, W=1); gx -> width
    const float Hdim = (float)LDIM;
    const float Wdim = 1.0f;
    const float base = -1.f + 2.f * (float)l / (float)(LDIM - 1);
    #pragma unroll
    for (int p = 0; p < NPOINT; ++p) {
      float gx = base + off[p] * 2.f / (float)(LDIM - 1);
      gx = fminf(1.f, fmaxf(-1.f, gx));
      const float gy  = 0.f;
      const float x   = (gx + 1.f) * 0.5f * (Wdim - 1.f);  // == 0 (W==1)
      const float y   = (gy + 1.f) * 0.5f * (Hdim - 1.f);  // == (L-1)/2
      const float x0  = floorf(x), y0f = floorf(y);
      const float wx  = x - x0;                            // == 0
      const float wy  = y - y0f;
      (void)wx;
      int y0 = (int)y0f;
      y0 = y0 < 0 ? 0 : (y0 > LDIM - 1 ? LDIM - 1 : y0);
      // x0i == x1i == 0 (clipped to W-1=0) -> x-lerp collapses exactly
      s_y0[h][p] = y0;
      s_wy[h][p] = wy;
      s_w [h][p] = e[p] * inv;
    }
  }
  __syncthreads();

  #pragma unroll
  for (int pass = 0; pass < DDIM / 256; ++pass) {
    const int d = pass * 256 + tid;
    const int h = d >> 6;              // d_k = 64
    float acc = 0.f;
    #pragma unroll
    for (int p = 0; p < NPOINT; ++p) {
      const int   y0 = s_y0[h][p];
      const int   y1 = (y0 + 1 > LDIM - 1) ? (LDIM - 1) : (y0 + 1);
      const float wy = s_wy[h][p];
      const float v0 = vproj[((size_t)b * LDIM + y0) * DDIM + d];
      const float v1 = vproj[((size_t)b * LDIM + y1) * DDIM + d];
      acc += s_w[h][p] * (v0 * (1.f - wy) + v1 * wy);
    }
    agg[(size_t)bl * DDIM + d] = acc;
  }
}

// ---------------------------------------------------------------------------
extern "C" void kernel_launch(void* const* d_in, const int* in_sizes, int n_in,
                              void* d_out, int out_size, void* d_ws, size_t ws_size,
                              hipStream_t stream) {
  const float* query = (const float*)d_in[0];
  // d_in[1] = key_in: unused by the reference
  const float* value = (const float*)d_in[2];
  const float* Wv    = (const float*)d_in[3];
  const float* bv    = (const float*)d_in[4];
  const float* Woff  = (const float*)d_in[5];
  const float* boff  = (const float*)d_in[6];
  const float* Ww    = (const float*)d_in[7];
  const float* bw    = (const float*)d_in[8];
  const float* Wout  = (const float*)d_in[9];
  const float* bout  = (const float*)d_in[10];
  float* out = (float*)d_out;

  // workspace layout (fp32): vproj | agg | poff | plogit
  char* ws = (char*)d_ws;
  float* vproj  = (float*)(ws);
  float* agg    = (float*)(ws + (size_t)MROWS * DDIM * 4);
  float* poff   = (float*)(ws + (size_t)MROWS * DDIM * 8);
  float* plogit = (float*)(ws + (size_t)MROWS * DDIM * 8 + (size_t)MROWS * HP * 4);

  const dim3 blk(256);

  // 1. value projection: 128x128 block tile, wave 32x64
  wmma_gemm_bias<128, 128, 2, 4>
      <<<dim3(MROWS / 128, DDIM / 128), blk, 0, stream>>>(
          value, Wv, bv, vproj, MROWS, DDIM, DDIM);
  // 2. offsets projection: 256x64 block tile, wave 64x32
  wmma_gemm_bias<256, 64, 4, 2>
      <<<dim3(MROWS / 256, HP / 64), blk, 0, stream>>>(
          query, Woff, boff, poff, MROWS, HP, DDIM);
  // 3. attention-logit projection
  wmma_gemm_bias<256, 64, 4, 2>
      <<<dim3(MROWS / 256, HP / 64), blk, 0, stream>>>(
          query, Ww, bw, plogit, MROWS, HP, DDIM);
  // 4. softmax + sample + aggregate
  sample_agg_kernel<<<dim3(MROWS), blk, 0, stream>>>(vproj, poff, plogit, agg);
  // 5. output projection
  wmma_gemm_bias<128, 128, 2, 4>
      <<<dim3(MROWS / 128, DDIM / 128), blk, 0, stream>>>(
          agg, Wout, bout, out, MROWS, DDIM, DDIM);
}
